// GNNEncoderAdjCou_73306501808320
// MI455X (gfx1250) — matrix operand order repaired and checked by reading, not compiled, not run on hardware
//
#include <hip/hip_runtime.h>
#include <hip/hip_bf16.h>
#include <math.h>

typedef __bf16 bf16_t;
typedef __attribute__((ext_vector_type(16))) __bf16 v16bf;
typedef __attribute__((ext_vector_type(8)))  __bf16 v8bf;
typedef __attribute__((ext_vector_type(8)))  float  v8f;
typedef __attribute__((ext_vector_type(4)))  unsigned int v4u;
typedef __attribute__((ext_vector_type(8)))  int v8i;
typedef __attribute__((ext_vector_type(4)))  int v4i;

#define BATCH 512
#define NNODE 128
#define EMB   64
#define NLAY  3
#define DCAT  192   // 3 * 64 concat

// LDS pitches (in elements)
#define PADJ 136    // 128 + 8 pad (bf16)
#define PH    72    // 64 + 8 pad  (bf16)
#define PHT  136    // 128 + 8 pad (bf16)
#define PW    72
#define PGP  200    // 192 + 8 pad (f32)

// dynamic-LDS byte offsets (all 16B aligned)
enum : unsigned {
  OFF_ADJ = 0u,                       // 128*136*2 = 34816
  OFF_H0  = OFF_ADJ + 128u*PADJ*2u,   // 34816
  OFF_H0T = OFF_H0  + 128u*PH*2u,     // 53248
  OFF_HA  = OFF_H0T + 64u*PHT*2u,     // 70656
  OFF_HB  = OFF_HA  + 128u*PH*2u,     // 89088
  OFF_HTA = OFF_HB  + 128u*PH*2u,     // 107520
  OFF_HTB = OFF_HTA + 64u*PHT*2u,     // 124928
  OFF_AGG = OFF_HTB + 64u*PHT*2u,     // 142336
  OFF_WT  = OFF_AGG + 128u*PH*2u,     // 160768
  OFF_WST = OFF_WT  + 64u*PW*2u,      // 169984
  OFF_BIA = OFF_WST + 64u*PW*2u,      // 179200
  OFF_GLO = OFF_BIA + 64u*4u,         // 179456
  OFF_MSK = OFF_GLO + 16u*PGP*4u,     // 192256
  OFF_STG = OFF_MSK + 128u*4u,        // 192768  (f32 TDM staging, 64KB)
  LDS_BYTES = OFF_STG + 128u*128u*4u  // 258304
};

// ---- WMMA fragment helpers (CDNA5 16-bit layouts, wave32) -------------------

// A 16x32: lane<16 -> K {0..7, 16..23}; lane>=16 -> K {8..15, 24..31}; M = lane&15
__device__ __forceinline__ v16bf load_a(const bf16_t* base, int pitch, int lane) {
  const int m = lane & 15, hsl = lane >> 4;
  const bf16_t* p = base + m * pitch + hsl * 8;
  v8bf lo = *(const v8bf*)(p);
  v8bf hi = *(const v8bf*)(p + 16);
  return __builtin_shufflevector(lo, hi, 0,1,2,3,4,5,6,7,8,9,10,11,12,13,14,15);
}

// B 32x16: lane<16 -> K {0..7, 16..23}; lane>=16 -> K {8..15, 24..31}; N = lane&15.
// Stored transposed in LDS as Bt[N][K] so each lane reads two 16-byte chunks.
__device__ __forceinline__ v16bf load_b(const bf16_t* base, int pitch, int lane) {
  const int n = lane & 15, hsl = lane >> 4;
  const bf16_t* p = base + n * pitch + hsl * 8;
  v8bf lo = *(const v8bf*)(p);
  v8bf hi = *(const v8bf*)(p + 16);
  return __builtin_shufflevector(lo, hi, 0,1,2,3,4,5,6,7,8,9,10,11,12,13,14,15);
}

// D 16x16 f32: VGPR r holds (M = r + 8*(lane>=16), N = lane&15)
__device__ __forceinline__ void store_rm(bf16_t* base, int pitch, int lane, v8f d) {
  const int n = lane & 15, hsl = lane >> 4;
  #pragma unroll
  for (int r = 0; r < 8; ++r)
    base[(hsl * 8 + r) * pitch + n] = (bf16_t)d[r];
}

// transposed store: 8 consecutive bf16 (16B) per lane into Bt layout
__device__ __forceinline__ void store_tr(bf16_t* base, int pitch, int lane, v8f d) {
  const int n = lane & 15, hsl = lane >> 4;
  v8bf v;
  #pragma unroll
  for (int r = 0; r < 8; ++r) v[r] = (bf16_t)d[r];
  *(v8bf*)(base + n * pitch + hsl * 8) = v;
}

__device__ __forceinline__ float selu_f(float x) {
  const float sc = 1.0507009873554805f;
  const float sa = 1.7580993408473766f; // scale * alpha
  return x > 0.f ? sc * x : sa * (__expf(x) - 1.f);
}

// ---- TDM: async DMA of one 128x128 f32 adjacency slab into LDS staging ------
// D# per CDNA5 ISA: group0 = {flags, lds_addr,
// global_addr[31:0], global_addr[56:32] | type=2<<30}; group1 packs
// data_size=4B, tensor_dim0/1 = 128, tile_dim0/1 = 128, dim0_stride = 128,
// dim1_stride = 16384. Groups 2/3 zero (2D tile). Tracked by TENSORcnt.
__device__ __forceinline__ void tdm_load_adj(const float* gptr, unsigned lds_off) {
  const unsigned long long ga = (unsigned long long)(uintptr_t)gptr;
  v4u g0;
  g0.x = 1u;                                            // count=1, user desc
  g0.y = lds_off;                                       // LDS byte address
  g0.z = (unsigned)(ga & 0xffffffffull);                // global_addr[31:0]
  g0.w = (unsigned)((ga >> 32) & 0x1ffffffull) | (2u << 30); // [56:32] | type=2
  v8i g1;
  g1[0] = (int)(2u << 16);        // workgroup_mask=0, data_size=2 (4 bytes)
  g1[1] = (int)(128u << 16);      // atomic_barrier=0 | tensor_dim0.lo16 = 128
  g1[2] = (int)(128u << 16);      // tensor_dim0.hi16=0 | tensor_dim1.lo16 = 128
  g1[3] = (int)(128u << 16);      // tensor_dim1.hi16=0 | tile_dim0 = 128
  g1[4] = 128;                    // tile_dim1 = 128, tile_dim2 = 0
  g1[5] = 128;                    // tensor_dim0_stride.lo32 = 128
  g1[6] = (int)(16384u << 16);    // d0_stride.hi16=0 | d1_stride.lo16 = 16384
  g1[7] = 0;                      // tensor_dim1_stride.hi32 = 0
  const v4i z = {0, 0, 0, 0};
#if __clang_major__ >= 23
  v8i z8 = {0,0,0,0,0,0,0,0};
  __builtin_amdgcn_tensor_load_to_lds(g0, g1, z, z, z8, 0);
#else
  __builtin_amdgcn_tensor_load_to_lds(g0, g1, z, z, 0);
#endif
}

// ---- fused GNN encoder: 1 workgroup == 1 batch element ----------------------
__global__ __launch_bounds__(256)
void gnn_fused(const int* __restrict__ xg,
               const float* __restrict__ adj0, const float* __restrict__ adj1,
               const float* __restrict__ adj2, const float* __restrict__ adj3,
               const float* __restrict__ cou,
               const unsigned char* __restrict__ maskg,
               const float* __restrict__ embg,
               const float* __restrict__ Wg,  const float* __restrict__ bg,
               const float* __restrict__ Wsg, const float* __restrict__ bsg,
               float* __restrict__ out)
{
  extern __shared__ __align__(16) char smem[];
  bf16_t* adjb   = (bf16_t*)(smem + OFF_ADJ);
  bf16_t* h0     = (bf16_t*)(smem + OFF_H0);
  bf16_t* h0t    = (bf16_t*)(smem + OFF_H0T);
  bf16_t* hA     = (bf16_t*)(smem + OFF_HA);
  bf16_t* hB     = (bf16_t*)(smem + OFF_HB);
  bf16_t* htA    = (bf16_t*)(smem + OFF_HTA);
  bf16_t* htB    = (bf16_t*)(smem + OFF_HTB);
  bf16_t* agg    = (bf16_t*)(smem + OFF_AGG);
  bf16_t* Wt     = (bf16_t*)(smem + OFF_WT);
  bf16_t* Wst    = (bf16_t*)(smem + OFF_WST);
  float*  biasc  = (float*)(smem + OFF_BIA);
  float*  glop   = (float*)(smem + OFF_GLO);
  int*    maskb  = (int*)(smem + OFF_MSK);
  float*  stage  = (float*)(smem + OFF_STG);

  const int t    = threadIdx.x;
  const int lane = t & 31;
  const int w    = t >> 5;        // wave id: owns output rows 16w..16w+15
  const int nn   = lane & 15;
  const int hs   = lane >> 4;
  const int bidx = blockIdx.x;

  // LDS byte offset of the staging buffer (flat LDS addr, low 32 bits)
  const unsigned stage_lds = (unsigned)(uintptr_t)(smem + OFF_STG);

  // ---- kick off DMA of bond 0 adjacency, gather embeddings + mask ----------
  if (t < 32) {
    tdm_load_adj(adj0 + (size_t)bidx * NNODE * NNODE, stage_lds);
  }
  if (t < NNODE) {
    const int   a = xg[bidx * NNODE + t];
    maskb[t] = (int)maskg[bidx * NNODE + t];
    const float* e = embg + (size_t)a * EMB;
    #pragma unroll 8
    for (int d = 0; d < EMB; ++d) {
      const float v = e[d];
      h0 [t * PH  + d] = (bf16_t)v;
      h0t[d * PHT + t] = (bf16_t)v;
    }
  }
  if (t < 32) __builtin_amdgcn_s_wait_tensorcnt(0);
  __syncthreads();                       // stage[bond0] + h0/mask visible

  float accA[NLAY][4][8];                // per-lane Sum_{adj bonds} selu(h_l)
  #pragma unroll
  for (int l = 0; l < NLAY; ++l)
    #pragma unroll
    for (int n = 0; n < 4; ++n)
      #pragma unroll
      for (int r = 0; r < 8; ++r) accA[l][n][r] = 0.f;

  float* out_glo = out;                                  // [512,192]
  float* out_loc = out + (size_t)BATCH * DCAT;           // [512*128,192]

  for (int bond = 0; bond < 5; ++bond) {
    // ---- convert staged f32 adjacency -> bf16 adjb (LDS -> LDS) ------------
    #pragma unroll 4
    for (int i = 0; i < 16; ++i) {
      const int lin = t + 256 * i;                 // float4 index, 4096 total
      const float4 v = ((const float4*)stage)[lin];
      const int row = lin >> 5;                    // 32 float4 per row
      const int c4  = (lin & 31) << 2;
      bf16_t* dst = adjb + row * PADJ + c4;
      dst[0] = (bf16_t)v.x; dst[1] = (bf16_t)v.y;
      dst[2] = (bf16_t)v.z; dst[3] = (bf16_t)v.w;
    }
    __syncthreads();                               // stage free for reuse

    // ---- overlap: DMA next bond's adjacency while we compute this one ------
    if (t < 32 && bond < 4) {
      const float* An = (bond == 0) ? adj1 : (bond == 1) ? adj2 :
                        (bond == 2) ? adj3 : cou;
      tdm_load_adj(An + (size_t)bidx * NNODE * NNODE, stage_lds);
    }

    const bf16_t* curh  = h0;
    const bf16_t* curht = h0t;

    #pragma unroll
    for (int l = 0; l < NLAY; ++l) {
      // ---- stage W^T, Wskip^T (bf16) and combined bias (f32) ----------------
      const size_t wbase = (size_t)(bond * NLAY + l) * EMB * EMB;
      const float* wp  = Wg  + wbase;
      const float* wsp = Wsg + wbase;
      for (int i = t; i < EMB * EMB; i += 256) {
        const int ki = i >> 6, ko = i & 63;        // W[in][out] -> Wt[out][in]
        Wt [ko * PW + ki] = (bf16_t)wp[i];
        Wst[ko * PW + ki] = (bf16_t)wsp[i];
      }
      if (t < EMB)
        biasc[t] = bg[(bond * NLAY + l) * EMB + t] + bsg[(bond * NLAY + l) * EMB + t];
      __syncthreads();

      // ---- matmul1: agg[16w..16w+15][0..63] = adj @ h  (K = 128) ------------
      {
        v8f acc[4] = {v8f{0.f}, v8f{0.f}, v8f{0.f}, v8f{0.f}};
        #pragma unroll
        for (int kt = 0; kt < 4; ++kt) {
          const v16bf a = load_a(adjb + w * 16 * PADJ + kt * 32, PADJ, lane);
          v16bf bfr[4];
          #pragma unroll
          for (int n = 0; n < 4; ++n)              // clause of 8 ds_load_b128
            bfr[n] = load_b(curht + n * 16 * PHT + kt * 32, PHT, lane);
          #pragma unroll
          for (int n = 0; n < 4; ++n)              // 4 back-to-back WMMAs
            acc[n] = __builtin_amdgcn_wmma_f32_16x16x32_bf16(
                         false, a, false, bfr[n], (short)0, acc[n], false, false);
        }
        #pragma unroll
        for (int n = 0; n < 4; ++n)
          store_rm(agg + w * 16 * PH + n * 16, PH, lane, acc[n]);
        // agg rows are wave-local: in-order LDS within a wave, no barrier
      }

      // ---- matmul2: h_new = agg @ W + h @ Wskip + bias  (K = 64) ------------
      v8f o[4] = {v8f{0.f}, v8f{0.f}, v8f{0.f}, v8f{0.f}};
      #pragma unroll
      for (int kt = 0; kt < 2; ++kt) {
        const v16bf aA = load_a(agg  + w * 16 * PH + kt * 32, PH, lane);
        const v16bf aS = load_a(curh + w * 16 * PH + kt * 32, PH, lane);
        v16bf bw[4];
        #pragma unroll
        for (int n = 0; n < 4; ++n)
          bw[n] = load_b(Wt + n * 16 * PW + kt * 32, PW, lane);
        #pragma unroll
        for (int n = 0; n < 4; ++n)
          o[n] = __builtin_amdgcn_wmma_f32_16x16x32_bf16(
                     false, aA, false, bw[n], (short)0, o[n], false, false);
        #pragma unroll
        for (int n = 0; n < 4; ++n)
          bw[n] = load_b(Wst + n * 16 * PW + kt * 32, PW, lane);
        #pragma unroll
        for (int n = 0; n < 4; ++n)
          o[n] = __builtin_amdgcn_wmma_f32_16x16x32_bf16(
                     false, aS, false, bw[n], (short)0, o[n], false, false);
      }
      #pragma unroll
      for (int n = 0; n < 4; ++n) {
        const float bv = biasc[n * 16 + nn];
        #pragma unroll
        for (int r = 0; r < 8; ++r) o[n][r] += bv;
      }

      // ---- write h_new (pre-activation feeds next layer) --------------------
      bf16_t* nh  = (l & 1) ? hB  : hA;
      bf16_t* nht = (l & 1) ? htB : htA;
      #pragma unroll
      for (int n = 0; n < 4; ++n) {
        store_rm(nh  + w * 16 * PH  + n * 16, PH,  lane, o[n]);
        store_tr(nht + n * 16 * PHT + w * 16, PHT, lane, o[n]);
      }

      if (bond < 4) {
        // accumulate selu(h_l) over the 4 adjacency bonds, in registers
        #pragma unroll
        for (int n = 0; n < 4; ++n)
          #pragma unroll
          for (int r = 0; r < 8; ++r)
            accA[l][n][r] += selu_f(o[n][r]);
      } else {
        // coulomb bond: fuse a + c - a*c, mask, emit loc + column partials
        #pragma unroll
        for (int n = 0; n < 4; ++n) {
          const int Dc = l * 64 + n * 16 + nn;
          float colsum = 0.f;
          #pragma unroll
          for (int r = 0; r < 8; ++r) {
            const int   row = w * 16 + hs * 8 + r;
            const float av  = accA[l][n][r];
            const float cv  = selu_f(o[n][r]);
            const float mol = av + cv - av * cv;
            const bool  mk  = maskb[row] != 0;
            out_loc[((size_t)bidx * NNODE + row) * DCAT + Dc] =
                mk ? mol : __int_as_float(0x7fc00000);
            colsum += mk ? mol : 0.f;
          }
          glop[(2 * w + hs) * PGP + Dc] = colsum;   // unique (slot,Dc) per lane
        }
      }
      __syncthreads();            // h_new/ht visible; Wt safe to overwrite
      curh  = nh;
      curht = nht;
    }

    // ---- bond boundary: make sure next adjacency DMA has landed ------------
    if (t < 32 && bond < 4) __builtin_amdgcn_s_wait_tensorcnt(0);
    __syncthreads();
  }

  // ---- masked global sum: reduce 16 per-(wave,half) partials ----------------
  if (t < DCAT) {
    float s = 0.f;
    #pragma unroll
    for (int i = 0; i < 16; ++i) s += glop[i * PGP + t];
    out_glo[(size_t)bidx * DCAT + t] = s;
  }
}

extern "C" void kernel_launch(void* const* d_in, const int* in_sizes, int n_in,
                              void* d_out, int out_size, void* d_ws, size_t ws_size,
                              hipStream_t stream) {
  (void)in_sizes; (void)n_in; (void)out_size; (void)d_ws; (void)ws_size;
  const int*           x    = (const int*)d_in[0];
  const float*         a0   = (const float*)d_in[1];
  const float*         a1   = (const float*)d_in[2];
  const float*         a2   = (const float*)d_in[3];
  const float*         a3   = (const float*)d_in[4];
  const float*         co   = (const float*)d_in[5];
  const unsigned char* mk   = (const unsigned char*)d_in[6];
  const float*         emb  = (const float*)d_in[7];
  const float*         W    = (const float*)d_in[8];
  const float*         bb   = (const float*)d_in[9];
  const float*         Ws   = (const float*)d_in[10];
  const float*         bs   = (const float*)d_in[11];
  float* out = (float*)d_out;

  hipLaunchKernelGGL(gnn_fused, dim3(BATCH), dim3(256), LDS_BYTES, stream,
                     x, a0, a1, a2, a3, co, mk, emb, W, bb, Ws, bs, out);
}
